// _SwinBlock_17678085391003
// MI455X (gfx1250) — matrix operand order repaired and checked by reading, not compiled, not run on hardware
//
#include <hip/hip_runtime.h>
#include <math.h>

// ---------------------------------------------------------------------------
// Swin block forward for MI455X (gfx1250, wave32, WMMA bf16 16x16x32).
// Async global->LDS staging (ASYNCcnt) with double-buffered LDS tiles.
// Workspace requirement: ~274 MB (weights-bf16 + token buffers + residual).
// ---------------------------------------------------------------------------

typedef __bf16 bf16;
typedef __attribute__((ext_vector_type(16))) __bf16 v16bf;
typedef __attribute__((ext_vector_type(8)))  __bf16 v8bf;
typedef __attribute__((ext_vector_type(8)))  float  v8f;
typedef int v4i __attribute__((vector_size(16)));

#define DIMC   384
#define NHEAD  12
#define HDIM   32
#define NTOK   49
#define MLPH   1536
#define HWPIX  3136
#define MTOT   50176   // 1024 windows * 49 tokens == 16*56*56 pixels
#define LDA    40      // padded LDS row stride (32 + 8 bf16)

#if defined(__gfx1250__) && __has_builtin(__builtin_amdgcn_global_load_async_to_lds_b128) && __has_builtin(__builtin_amdgcn_s_wait_asynccnt)
#define HAS_ASYNC_LDS 1
#pragma message("gfx1250: using GLOBAL_LOAD_ASYNC_TO_LDS_B128 + s_wait_asynccnt")
#else
#define HAS_ASYNC_LDS 0
#pragma message("gfx1250: async-to-LDS builtins NOT available; sync staging fallback")
#endif

// 16-byte global -> LDS copy (async on CDNA5, sync fallback otherwise)
static __device__ __forceinline__ void copy16_g2l(const void* g, void* l) {
#if HAS_ASYNC_LDS
  __builtin_amdgcn_global_load_async_to_lds_b128(
      (__attribute__((address_space(1))) v4i*)g,
      (__attribute__((address_space(3))) v4i*)l, 0, 0);
#else
  *(v8bf*)l = *(const v8bf*)g;
#endif
}
static __device__ __forceinline__ void g2l_wait_all() {
#if HAS_ASYNC_LDS
  __builtin_amdgcn_s_wait_asynccnt(0);
#endif
}

static __device__ __forceinline__ v16bf make_v16(v8bf lo, v8bf hi) {
  union { v16bf v; v8bf h[2]; } u; u.h[0] = lo; u.h[1] = hi; return u.v;
}
static __device__ __forceinline__ v8f wmma_bf16(v16bf a, v16bf b, v8f c) {
  return __builtin_amdgcn_wmma_f32_16x16x32_bf16(false, a, false, b, (short)0, c,
                                                 false, false);
}
static __device__ __forceinline__ v8f v8f_zero() {
  v8f z = {0.f, 0.f, 0.f, 0.f, 0.f, 0.f, 0.f, 0.f};
  return z;
}

// token (window order) + channel -> flat index in (B,C,56,56) with un-shift
static __device__ __forceinline__ size_t win_tok_to_img(long gm, int ch) {
  long b_ = gm / 49; int t = (int)(gm % 49);
  int bb = (int)(b_ >> 6); int wi = (int)(b_ & 63);
  int wh = wi >> 3, ww = wi & 7;
  int rr = t / 7, cc = t % 7;
  int oh = (wh * 7 + rr + 3) % 56;
  int ow = (ww * 7 + cc + 3) % 56;
  return ((size_t)bb * DIMC + ch) * HWPIX + oh * 56 + ow;
}

static __device__ __forceinline__ int regidx(int a) {
  return a < 49 ? 0 : (a < 53 ? 1 : 2);
}

// ---------------------------------------------------------------------------
// Stage one 64x32 A chunk + 128x32 B chunk into LDS (256 threads).
// ---------------------------------------------------------------------------
static __device__ __forceinline__ void stage_chunk(
    const bf16* __restrict__ A, const bf16* __restrict__ Bt, int K,
    long m0, long n0, int k0, bf16* As, bf16* Bs, int tid) {
  {
    int row = tid >> 2, cg = (tid & 3) * 8;
    copy16_g2l(A + (m0 + row) * (long)K + k0 + cg, As + row * LDA + cg);
  }
#pragma unroll
  for (int j = 0; j < 2; ++j) {
    int vid = tid + j * 256;
    int n = vid >> 2, cg = (vid & 3) * 8;
    copy16_g2l(Bt + (n0 + n) * (long)K + k0 + cg, Bs + n * LDA + cg);
  }
}

// ---------------------------------------------------------------------------
// Generic 64x128 bf16 GEMM core: C(64x128) += A(64xK) * Bt(N x K)^T
// Block: 256 threads = 8 waves laid out 4(m) x 2(n); each wave: 16 x 64.
// LDS double-buffered; async copy of chunk k+1 overlaps WMMA of chunk k.
// ---------------------------------------------------------------------------
static __device__ __forceinline__ void gemm_core_64x128(
    const bf16* __restrict__ A, const bf16* __restrict__ Bt, int K,
    long m0, long n0, bf16* As, bf16* Bs, v8f acc[4]) {
  const int tid = threadIdx.x;
  const int wave = tid >> 5, lane = tid & 31;
  const int wm = wave >> 1, wn = wave & 1;
  const int lhalf = lane >> 4, l16 = lane & 15;
#pragma unroll
  for (int s = 0; s < 4; ++s) acc[s] = v8f_zero();

  const int nk = K >> 5;
  stage_chunk(A, Bt, K, m0, n0, 0, As, Bs, tid);

  for (int it = 0; it < nk; ++it) {
    bf16* Ac = As + (it & 1) * (64 * LDA);
    bf16* Bc = Bs + (it & 1) * (128 * LDA);
    g2l_wait_all();      // my async copies into current buffer done
    __syncthreads();     // everyone's copies done; prev-iter reads done
    if (it + 1 < nk)     // overlap: fill other buffer while we do WMMAs
      stage_chunk(A, Bt, K, m0, n0, (it + 1) * 32,
                  As + ((it + 1) & 1) * (64 * LDA),
                  Bs + ((it + 1) & 1) * (128 * LDA), tid);

    // A fragment: lane l -> row (l&15), K halves (l>>4)*8 and +16
    int am = wm * 16 + l16;
    int kb = lhalf * 8;
    v16bf afrag = make_v16(*(const v8bf*)(Ac + am * LDA + kb),
                           *(const v8bf*)(Ac + am * LDA + kb + 16));
#pragma unroll
    for (int s = 0; s < 4; ++s) {
      // B fragment: lane l -> column (l&15), K halves (l>>4)*8 and +16
      int bn = wn * 64 + s * 16 + l16;
      int kb2 = lhalf * 8;
      v16bf bfrag = make_v16(*(const v8bf*)(Bc + bn * LDA + kb2),
                             *(const v8bf*)(Bc + bn * LDA + kb2 + 16));
      acc[s] = wmma_bf16(afrag, bfrag, acc[s]);
    }
  }
}

// ---------------------------------------------------------------------------
// Weight convert + transpose: src fp32 [K][N] -> dst bf16 [N][K]
// ---------------------------------------------------------------------------
__global__ void k_cvt_t(const float* __restrict__ src, bf16* __restrict__ dst,
                        int K, int N) {
  long i = (long)blockIdx.x * 256 + threadIdx.x;
  if (i >= (long)N * K) return;
  long n = i / K, kk = i % K;
  dst[i] = (bf16)src[kk * (long)N + n];
}

// ---------------------------------------------------------------------------
// LayerNorm over channels; windowed=1: LN1 + roll(-3,-3) + window partition.
// src layout (B,C,56,56) fp32; dst token-major bf16 [token][384].
// Block: 384 threads (one per channel), grid: 50176 tokens.
// ---------------------------------------------------------------------------
__global__ void k_ln(const float* __restrict__ src, const float* __restrict__ g,
                     const float* __restrict__ beta, bf16* __restrict__ dst,
                     int windowed) {
  int blk = blockIdx.x;
  int ch = threadIdx.x;
  int bb, sh, sw;
  if (windowed) {
    int b_ = blk / 49, t = blk % 49;
    bb = b_ >> 6; int wi = b_ & 63;
    int wh = wi >> 3, ww = wi & 7;
    int rr = t / 7, cc = t % 7;
    sh = (wh * 7 + rr + 3) % 56;   // rolled(-3) gather
    sw = (ww * 7 + cc + 3) % 56;
  } else {
    bb = blk / HWPIX; int p = blk % HWPIX;
    sh = p / 56; sw = p % 56;
  }
  float val = src[((size_t)bb * DIMC + ch) * HWPIX + sh * 56 + sw];

  __shared__ float red[14];
  float s = val;
#pragma unroll
  for (int o = 16; o > 0; o >>= 1) s += __shfl_down(s, o, 32);
  if ((ch & 31) == 0) red[ch >> 5] = s;
  __syncthreads();
  if (ch == 0) {
    float t2 = 0.f;
    for (int i = 0; i < 12; ++i) t2 += red[i];
    red[12] = t2 * (1.f / 384.f);
  }
  __syncthreads();
  float mu = red[12];
  float dv = val - mu;
  s = dv * dv;
#pragma unroll
  for (int o = 16; o > 0; o >>= 1) s += __shfl_down(s, o, 32);
  if ((ch & 31) == 0) red[ch >> 5] = s;
  __syncthreads();
  if (ch == 0) {
    float t2 = 0.f;
    for (int i = 0; i < 12; ++i) t2 += red[i];
    red[13] = t2 * (1.f / 384.f);
  }
  __syncthreads();
  float var = red[13];
  float y = dv * rsqrtf(var + 1e-5f) * g[ch] + beta[ch];
  dst[(size_t)blk * DIMC + ch] = (bf16)y;
}

// ---------------------------------------------------------------------------
// QKV GEMM: wins(M x 384) @ qkv_wT -> q,k,v in (B_,NH,49,32) bf16; q pre-scaled
// ---------------------------------------------------------------------------
__global__ void k_gemm_qkv(const bf16* __restrict__ A, const bf16* __restrict__ Bt,
                           const float* __restrict__ bias, bf16* __restrict__ q,
                           bf16* __restrict__ k, bf16* __restrict__ v) {
  __shared__ __align__(16) bf16 As[2 * 64 * LDA];
  __shared__ __align__(16) bf16 Bs[2 * 128 * LDA];
  v8f acc[4];
  long m0 = (long)blockIdx.x * 64;
  long n0 = (long)blockIdx.y * 128;
  gemm_core_64x128(A, Bt, DIMC, m0, n0, As, Bs, acc);

  const int tid = threadIdx.x, wave = tid >> 5, lane = tid & 31;
  const int wm = wave >> 1, wn = wave & 1, lhalf = lane >> 4, l16 = lane & 15;
  const float scale = 0.1767766952966369f;  // 1/sqrt(32)
#pragma unroll
  for (int s = 0; s < 4; ++s) {
    int gn = (int)n0 + wn * 64 + s * 16 + l16;
    int s3 = gn / DIMC, rem = gn % DIMC;
    int h = rem >> 5, d = rem & 31;
    bf16* dst = (s3 == 0) ? q : ((s3 == 1) ? k : v);
    float bi = bias[gn];
#pragma unroll
    for (int r = 0; r < 8; ++r) {
      long gm = m0 + wm * 16 + lhalf * 8 + r;
      float val = acc[s][r] + bi;
      if (s3 == 0) val *= scale;
      long b_ = gm / 49, t = gm % 49;
      dst[((b_ * NHEAD + h) * NTOK + t) * HDIM + d] = (bf16)val;
    }
  }
}

// ---------------------------------------------------------------------------
// Attention: one block per (window, head). 128 threads = 4 waves (one m-tile).
// scores = q@k^T (scale folded in q) + rel-pos bias + shift mask -> softmax
// -> out = P@v, stored token-major (B_,49,384).
// ---------------------------------------------------------------------------
__global__ void k_attn(const bf16* __restrict__ q, const bf16* __restrict__ k,
                       const bf16* __restrict__ v, const float* __restrict__ relb,
                       bf16* __restrict__ out) {
  __shared__ __align__(16) bf16 qs[64 * LDA];
  __shared__ __align__(16) bf16 ks[64 * LDA];
  __shared__ __align__(16) bf16 vT[32 * 72];
  __shared__ float sc[64 * 65];
  __shared__ __align__(16) bf16 ps[64 * 72];

  const int blk = blockIdx.x;
  const int b_ = blk / NHEAD, h = blk % NHEAD;
  const int tid = threadIdx.x;

  const size_t base = ((size_t)b_ * NHEAD + h) * NTOK * HDIM;
  const unsigned short* vsrc = (const unsigned short*)(v + base);
  unsigned short* qsd = (unsigned short*)qs;
  unsigned short* ksd = (unsigned short*)ks;
  unsigned short* vtd = (unsigned short*)vT;

  // zero-pad rows >= 49 of q/k and build transposed vT (scalar gather)
  for (int i = tid; i < 64 * 32; i += 128) {
    int t = i >> 5, d = i & 31;
    if (t >= NTOK) {
      qsd[t * LDA + d] = 0;
      ksd[t * LDA + d] = 0;
    }
    vtd[d * 72 + t] = (t < NTOK) ? vsrc[t * 32 + d] : (unsigned short)0;
  }
  // async-copy valid q/k rows (49 rows x 64B = 4 chunks of 16B each)
  for (int idx = tid; idx < NTOK * 4; idx += 128) {
    int t = idx >> 2, cg = (idx & 3) * 8;
    copy16_g2l(q + base + t * 32 + cg, qs + t * LDA + cg);
    copy16_g2l(k + base + t * 32 + cg, ks + t * LDA + cg);
  }
  g2l_wait_all();
  __syncthreads();

  const int wave = tid >> 5, lane = tid & 31;
  const int lhalf = lane >> 4, l16 = lane & 15;
  const int wm = wave;  // 4 waves -> rows 0..63

  // phase 1: scores (K = HD = 32, single WMMA step)
  {
    int am = wm * 16 + l16;
    int kb = lhalf * 8;
    v16bf a = make_v16(*(const v8bf*)(qs + am * LDA + kb),
                       *(const v8bf*)(qs + am * LDA + kb + 16));
#pragma unroll
    for (int s = 0; s < 4; ++s) {
      int bn = s * 16 + l16;
      int kb2 = lhalf * 8;
      v16bf b = make_v16(*(const v8bf*)(ks + bn * LDA + kb2),
                         *(const v8bf*)(ks + bn * LDA + kb2 + 16));
      v8f c = v8f_zero();
      c = wmma_bf16(a, b, c);
#pragma unroll
      for (int r = 0; r < 8; ++r)
        sc[(wm * 16 + lhalf * 8 + r) * 65 + s * 16 + l16] = c[r];
    }
  }
  __syncthreads();

  // phase 2: bias + mask + softmax (one thread per row)
  if (tid < 64) {
    int row = tid;
    if (row < NTOK) {
      int r7 = row / 7, rc = row % 7;
      int wi = b_ & 63;
      int wh = wi >> 3, ww = wi & 7;
      int labr = regidx(wh * 7 + r7) * 3 + regidx(ww * 7 + rc);
      float mx = -1e30f;
      for (int j = 0; j < NTOK; ++j) {
        int j7 = j / 7, jc = j % 7;
        int idx = (r7 - j7 + 6) * 13 + (rc - jc + 6);
        float bias = relb[idx * NHEAD + h];
        int labj = regidx(wh * 7 + j7) * 3 + regidx(ww * 7 + jc);
        float val = sc[row * 65 + j] + bias + ((labj == labr) ? 0.f : -100.f);
        sc[row * 65 + j] = val;
        mx = fmaxf(mx, val);
      }
      float sum = 0.f;
      for (int j = 0; j < NTOK; ++j) {
        float e = __expf(sc[row * 65 + j] - mx);
        sum += e;
        sc[row * 65 + j] = e;
      }
      float inv = 1.f / sum;
      for (int j = 0; j < 64; ++j)
        ps[row * 72 + j] = (j < NTOK) ? (bf16)(sc[row * 65 + j] * inv) : (bf16)0.f;
    } else {
      for (int j = 0; j < 64; ++j) ps[row * 72 + j] = (bf16)0.f;
    }
  }
  __syncthreads();

  // phase 3: O = P(64x64) @ V(64x32): 2 k-steps x 2 n-subtiles
  v8f o[2];
  o[0] = v8f_zero();
  o[1] = v8f_zero();
#pragma unroll
  for (int kst = 0; kst < 2; ++kst) {
    int am = wm * 16 + l16;
    int kb = kst * 32 + lhalf * 8;
    v16bf a = make_v16(*(const v8bf*)(ps + am * 72 + kb),
                       *(const v8bf*)(ps + am * 72 + kb + 16));
#pragma unroll
    for (int s = 0; s < 2; ++s) {
      int bn = s * 16 + l16;
      int kb2 = kst * 32 + lhalf * 8;
      v16bf b = make_v16(*(const v8bf*)(vT + bn * 72 + kb2),
                         *(const v8bf*)(vT + bn * 72 + kb2 + 16));
      o[s] = wmma_bf16(a, b, o[s]);
    }
  }
#pragma unroll
  for (int s = 0; s < 2; ++s)
#pragma unroll
    for (int r = 0; r < 8; ++r) {
      int m = wm * 16 + lhalf * 8 + r;
      if (m < NTOK)
        out[((size_t)b_ * NTOK + m) * DIMC + h * HDIM + s * 16 + l16] =
            (bf16)o[s][r];
    }
}

// ---------------------------------------------------------------------------
// Proj GEMM + window reverse + un-shift + residual into x_res (B,C,56,56) f32
// ---------------------------------------------------------------------------
__global__ void k_gemm_proj(const bf16* __restrict__ A, const bf16* __restrict__ Bt,
                            const float* __restrict__ bias,
                            const float* __restrict__ x_in,
                            float* __restrict__ x_res) {
  __shared__ __align__(16) bf16 As[2 * 64 * LDA];
  __shared__ __align__(16) bf16 Bs[2 * 128 * LDA];
  v8f acc[4];
  long m0 = (long)blockIdx.x * 64;
  long n0 = (long)blockIdx.y * 128;
  gemm_core_64x128(A, Bt, DIMC, m0, n0, As, Bs, acc);

  const int tid = threadIdx.x, wave = tid >> 5, lane = tid & 31;
  const int wm = wave >> 1, wn = wave & 1, lhalf = lane >> 4, l16 = lane & 15;
#pragma unroll
  for (int s = 0; s < 4; ++s) {
    int gn = (int)n0 + wn * 64 + s * 16 + l16;
    float bi = bias[gn];
#pragma unroll
    for (int r = 0; r < 8; ++r) {
      long gm = m0 + wm * 16 + lhalf * 8 + r;
      size_t idx = win_tok_to_img(gm, gn);
      x_res[idx] = x_in[idx] + acc[s][r] + bi;
    }
  }
}

// ---------------------------------------------------------------------------
// MLP GEMM1 + exact GELU -> hidden bf16 [token][1536]
// ---------------------------------------------------------------------------
__global__ void k_gemm_mlp1(const bf16* __restrict__ A, const bf16* __restrict__ Bt,
                            const float* __restrict__ bias,
                            bf16* __restrict__ hbuf) {
  __shared__ __align__(16) bf16 As[2 * 64 * LDA];
  __shared__ __align__(16) bf16 Bs[2 * 128 * LDA];
  v8f acc[4];
  long m0 = (long)blockIdx.x * 64;
  long n0 = (long)blockIdx.y * 128;
  gemm_core_64x128(A, Bt, DIMC, m0, n0, As, Bs, acc);

  const int tid = threadIdx.x, wave = tid >> 5, lane = tid & 31;
  const int wm = wave >> 1, wn = wave & 1, lhalf = lane >> 4, l16 = lane & 15;
#pragma unroll
  for (int s = 0; s < 4; ++s) {
    int gn = (int)n0 + wn * 64 + s * 16 + l16;
    float bi = bias[gn];
#pragma unroll
    for (int r = 0; r < 8; ++r) {
      long gm = m0 + wm * 16 + lhalf * 8 + r;
      float t = acc[s][r] + bi;
      float gl = 0.5f * t * (1.f + erff(t * 0.70710678118654752f));
      hbuf[gm * (long)MLPH + gn] = (bf16)gl;
    }
  }
}

// ---------------------------------------------------------------------------
// MLP GEMM2 + residual; scatter into d_out (B,C,56,56) f32 (raster tokens)
// ---------------------------------------------------------------------------
__global__ void k_gemm_mlp2(const bf16* __restrict__ A, const bf16* __restrict__ Bt,
                            const float* __restrict__ bias,
                            const float* __restrict__ x_res,
                            float* __restrict__ out) {
  __shared__ __align__(16) bf16 As[2 * 64 * LDA];
  __shared__ __align__(16) bf16 Bs[2 * 128 * LDA];
  v8f acc[4];
  long m0 = (long)blockIdx.x * 64;
  long n0 = (long)blockIdx.y * 128;
  gemm_core_64x128(A, Bt, MLPH, m0, n0, As, Bs, acc);

  const int tid = threadIdx.x, wave = tid >> 5, lane = tid & 31;
  const int wm = wave >> 1, wn = wave & 1, lhalf = lane >> 4, l16 = lane & 15;
#pragma unroll
  for (int s = 0; s < 4; ++s) {
    int gn = (int)n0 + wn * 64 + s * 16 + l16;
    float bi = bias[gn];
#pragma unroll
    for (int r = 0; r < 8; ++r) {
      long gm = m0 + wm * 16 + lhalf * 8 + r;  // raster token
      int bb = (int)(gm / HWPIX);
      int p = (int)(gm % HWPIX);
      int oh = p / 56, ow = p % 56;
      size_t idx = ((size_t)bb * DIMC + gn) * HWPIX + oh * 56 + ow;
      out[idx] = x_res[idx] + acc[s][r] + bi;
    }
  }
}

// ---------------------------------------------------------------------------
extern "C" void kernel_launch(void* const* d_in, const int* in_sizes, int n_in,
                              void* d_out, int out_size, void* d_ws, size_t ws_size,
                              hipStream_t stream) {
  (void)in_sizes; (void)n_in; (void)out_size; (void)ws_size;
  const float* x      = (const float*)d_in[0];
  const float* relb   = (const float*)d_in[1];
  const float* qkv_w  = (const float*)d_in[2];
  const float* qkv_b  = (const float*)d_in[3];
  const float* proj_w = (const float*)d_in[4];
  const float* proj_b = (const float*)d_in[5];
  const float* n1g    = (const float*)d_in[6];
  const float* n1b    = (const float*)d_in[7];
  const float* n2g    = (const float*)d_in[8];
  const float* n2b    = (const float*)d_in[9];
  const float* mlp_w1 = (const float*)d_in[10];
  const float* mlp_b1 = (const float*)d_in[11];
  const float* mlp_w2 = (const float*)d_in[12];
  const float* mlp_b2 = (const float*)d_in[13];
  float* out = (float*)d_out;

  char* ws = (char*)d_ws;
  size_t off = 0;
  bf16* wqkvT  = (bf16*)(ws + off); off += (size_t)1152 * 384 * 2;
  bf16* wprojT = (bf16*)(ws + off); off += (size_t)384 * 384 * 2;
  bf16* wm1T   = (bf16*)(ws + off); off += (size_t)1536 * 384 * 2;
  bf16* wm2T   = (bf16*)(ws + off); off += (size_t)384 * 1536 * 2;
  bf16* wins   = (bf16*)(ws + off); off += (size_t)MTOT * DIMC * 2;  // also xn2
  bf16* qbuf   = (bf16*)(ws + off); off += (size_t)MTOT * DIMC * 2;
  bf16* kbuf   = (bf16*)(ws + off); off += (size_t)MTOT * DIMC * 2;
  bf16* vbuf   = (bf16*)(ws + off); off += (size_t)MTOT * DIMC * 2;
  bf16* abuf   = (bf16*)(ws + off); off += (size_t)MTOT * DIMC * 2;
  float* xres  = (float*)(ws + off); off += (size_t)MTOT * DIMC * 4;
  // MLP hidden (MTOT x 1536 bf16 = 154,140,672 B) reuses q..attn region
  bf16* hbuf = qbuf;

  // 1) weight convert + transpose (one-shot, tiny)
  k_cvt_t<<<dim3((1152 * 384 + 255) / 256), 256, 0, stream>>>(qkv_w, wqkvT, 384, 1152);
  k_cvt_t<<<dim3((384 * 384 + 255) / 256), 256, 0, stream>>>(proj_w, wprojT, 384, 384);
  k_cvt_t<<<dim3((384 * 1536 + 255) / 256), 256, 0, stream>>>(mlp_w1, wm1T, 384, 1536);
  k_cvt_t<<<dim3((1536 * 384 + 255) / 256), 256, 0, stream>>>(mlp_w2, wm2T, 1536, 384);

  // 2) LN1 + shift + window partition
  k_ln<<<dim3(MTOT), 384, 0, stream>>>(x, n1g, n1b, wins, 1);

  // 3) QKV GEMM (M=50176, K=384, N=1152)
  k_gemm_qkv<<<dim3(MTOT / 64, 1152 / 128), 256, 0, stream>>>(wins, wqkvT, qkv_b,
                                                              qbuf, kbuf, vbuf);

  // 4) windowed attention (1024 windows x 12 heads)
  k_attn<<<dim3(1024 * NHEAD), 128, 0, stream>>>(qbuf, kbuf, vbuf, relb, abuf);

  // 5) proj GEMM + window reverse + residual (N=384)
  k_gemm_proj<<<dim3(MTOT / 64, 384 / 128), 256, 0, stream>>>(abuf, wprojT, proj_b,
                                                              x, xres);

  // 6) LN2 (raster tokens)
  k_ln<<<dim3(MTOT), 384, 0, stream>>>(xres, n2g, n2b, wins, 0);

  // 7) MLP GEMM1 + GELU (N=1536)
  k_gemm_mlp1<<<dim3(MTOT / 64, 1536 / 128), 256, 0, stream>>>(wins, wm1T, mlp_b1,
                                                               hbuf);

  // 8) MLP GEMM2 + residual -> d_out (K=1536, N=384)
  k_gemm_mlp2<<<dim3(MTOT / 64, 384 / 128), 256, 0, stream>>>(hbuf, wm2T, mlp_b2,
                                                              xres, out);
}
